// MultiHeadAttention_34591666602159
// MI455X (gfx1250) — matrix operand order repaired and checked by reading, not compiled, not run on hardware
//
#include <hip/hip_runtime.h>

// ---------------------------------------------------------------------------
// MultiHeadAttention + output proj + LayerNorm for MI455X (gfx1250, wave32).
// All GEMMs via v_wmma_f32_16x16x32_bf16 (f32 accumulate), register-blocked
// 4x in M so each B-fragment load feeds 4 WMMAs (cuts L2 traffic ~4x).
// ---------------------------------------------------------------------------

#define NHEAD   16
#define DMODEL  1024
#define DK      64
#define DV      64
#define BB      4
#define SS      2048
#define ROWS    (BB * SS)          // 8192
#define MT      4                  // M-direction register blocking (4 x 16 rows)

typedef __attribute__((ext_vector_type(16))) __bf16 v16bf;
typedef __attribute__((ext_vector_type(8)))  float  v8f;

#define WMMA_BF16(a, b, c) \
  __builtin_amdgcn_wmma_f32_16x16x32_bf16(false, (a), false, (b), (short)0, (c), false, false)

__device__ __forceinline__ __bf16 f2bf(float f) { return (__bf16)f; }

__device__ __forceinline__ v8f zero8() {
  v8f z = {0.f, 0.f, 0.f, 0.f, 0.f, 0.f, 0.f, 0.f};
  return z;
}

union Frag16 { v16bf v; uint4 q[2]; };

// 16 contiguous bf16 (one 32B chunk) -> B fragment lane data
__device__ __forceinline__ v16bf load_frag_16(const __bf16* p) {
  Frag16 u;
  u.q[0] = ((const uint4*)p)[0];
  u.q[1] = ((const uint4*)p)[1];
  return u.v;
}

// two 8-bf16 chunks -> A fragment lane data (K = h*16+0..7 and h*16+8..15)
__device__ __forceinline__ v16bf load_frag_2x8(const __bf16* p0, const __bf16* p1) {
  Frag16 u;
  u.q[0] = ((const uint4*)p0)[0];
  u.q[1] = ((const uint4*)p1)[0];
  return u.v;
}

// ---------------------------------------------------------------------------
// K0a: elementwise fp32 -> bf16 (for x). 4 elements per thread.
// ---------------------------------------------------------------------------
__global__ __launch_bounds__(256)
void cvt_bf16_kernel(const float* __restrict__ src, __bf16* __restrict__ dst,
                     size_t total4) {
  size_t i = (size_t)blockIdx.x * 256 + threadIdx.x;
  if (i >= total4) return;
  const float4 f = ((const float4*)src)[i];
  union { __bf16 h[4]; unsigned long long u; } o;
  o.h[0] = f2bf(f.x); o.h[1] = f2bf(f.y); o.h[2] = f2bf(f.z); o.h[3] = f2bf(f.w);
  ((unsigned long long*)dst)[i] = o.u;
}

// ---------------------------------------------------------------------------
// K0b: transpose + convert fp32 [batch, R, C] -> bf16 [batch, C, R]
// ---------------------------------------------------------------------------
__global__ __launch_bounds__(256)
void transpose_cvt_kernel(const float* __restrict__ src, __bf16* __restrict__ dst,
                          int batch, int R, int C) {
  size_t i = (size_t)blockIdx.x * 256 + threadIdx.x;
  size_t total = (size_t)batch * R * C;
  if (i >= total) return;
  int r = (int)(i % R);
  size_t t = i / R;
  int c = (int)(t % C);
  int b = (int)(t / C);
  dst[i] = f2bf(src[((size_t)b * R + r) * C + c]);
}

// ---------------------------------------------------------------------------
// K1: QKV projection.  grid = (128 row-blocks of 64, 16 heads, 3 [q,k,v]).
// One wave computes a 64x64 tile of x @ W + b (16 accumulators).
// q scaled by 1/sqrt(Dk); v stored transposed [B,H,DV,S].
// ---------------------------------------------------------------------------
__global__ __launch_bounds__(32)
void qkv_proj_kernel(const __bf16* __restrict__ xbf,
                     const __bf16* __restrict__ WqT, const __bf16* __restrict__ WkT,
                     const __bf16* __restrict__ WvT,
                     const float* __restrict__ bq, const float* __restrict__ bk,
                     const float* __restrict__ bv,
                     __bf16* __restrict__ qout, __bf16* __restrict__ kout,
                     __bf16* __restrict__ vTout) {
  const int lane = threadIdx.x;
  const int half = lane >> 4;
  const int lr = lane & 15;
  const int rb = blockIdx.x;           // 64-row block
  const int h = blockIdx.y;
  const int z = blockIdx.z;            // 0=q 1=k 2=v

  const __bf16* WT = (z == 0) ? WqT : (z == 1) ? WkT : WvT;
  const float* bias = (z == 0) ? bq : (z == 1) ? bk : bv;

  v8f acc[MT][4];
#pragma unroll
  for (int mt = 0; mt < MT; ++mt)
#pragma unroll
    for (int nt = 0; nt < 4; ++nt) acc[mt][nt] = zero8();

  const __bf16* xrow0 = xbf + (size_t)(rb * 64 + lr) * DMODEL;

  for (int k0 = 0; k0 < DMODEL; k0 += 32) {
    v16bf a[MT];
#pragma unroll
    for (int mt = 0; mt < MT; ++mt) {
      const __bf16* xr = xrow0 + (size_t)mt * 16 * DMODEL + k0;
      a[mt] = load_frag_2x8(xr + half * 16, xr + half * 16 + 8);
    }
    const __bf16* wb = WT + (size_t)h * DK * DMODEL + k0 + half * 16;
    v16bf b0 = load_frag_16(wb + (size_t)(lr) * DMODEL);
    v16bf b1 = load_frag_16(wb + (size_t)(16 + lr) * DMODEL);
    v16bf b2 = load_frag_16(wb + (size_t)(32 + lr) * DMODEL);
    v16bf b3 = load_frag_16(wb + (size_t)(48 + lr) * DMODEL);
#pragma unroll
    for (int mt = 0; mt < MT; ++mt) {
      acc[mt][0] = WMMA_BF16(a[mt], b0, acc[mt][0]);
      acc[mt][1] = WMMA_BF16(a[mt], b1, acc[mt][1]);
      acc[mt][2] = WMMA_BF16(a[mt], b2, acc[mt][2]);
      acc[mt][3] = WMMA_BF16(a[mt], b3, acc[mt][3]);
    }
  }

#pragma unroll
  for (int mt = 0; mt < MT; ++mt) {
#pragma unroll
    for (int nt = 0; nt < 4; ++nt) {
#pragma unroll
      for (int j = 0; j < 8; ++j) {
        int m = j + 8 * half;
        int n = nt * 16 + lr;
        float val = acc[mt][nt][j] + bias[h * DK + n];
        if (z == 0) val *= 0.125f;     // 1/sqrt(64) folded into q
        int row = rb * 64 + mt * 16 + m;  // row = b*S + s
        int b = row >> 11;
        int s = row & (SS - 1);
        __bf16 bvv = f2bf(val);
        if (z == 2) {
          vTout[((((size_t)b * NHEAD + h) * DV) + n) * SS + s] = bvv;
        } else {
          __bf16* dst = (z == 0) ? qout : kout;
          dst[((((size_t)b * NHEAD + h) * SS) + s) * DK + n] = bvv;
        }
      }
    }
  }
}

// ---------------------------------------------------------------------------
// K2: flash attention. grid = (32 q-blocks of 64, 16 heads, 4 batch),
// 1 wave/block covering 64 queries (4 q-tiles). Per 32-key block:
// K/V fragments loaded once, 16 score WMMAs + 16 PV WMMAs.
// ---------------------------------------------------------------------------
__global__ __launch_bounds__(32)
void attn_kernel(const __bf16* __restrict__ qg, const __bf16* __restrict__ kg,
                 const __bf16* __restrict__ vTg, __bf16* __restrict__ cat) {
  __shared__ float stile[16][32];
  const int lane = threadIdx.x;
  const int half = lane >> 4;
  const int lr = lane & 15;
  const int qb = blockIdx.x;           // 64-query block
  const int h = blockIdx.y;
  const int b = blockIdx.z;

  const __bf16* qbase = qg + (((size_t)b * NHEAD + h) * SS) * DK;
  const __bf16* kbase = kg + (((size_t)b * NHEAD + h) * SS) * DK;
  const __bf16* vbase = vTg + (((size_t)b * NHEAD + h) * DV) * SS;

  v16bf aq[MT][2];
#pragma unroll
  for (int qt = 0; qt < MT; ++qt) {
    const __bf16* qrow = qbase + (size_t)(qb * 64 + qt * 16 + lr) * DK;
    aq[qt][0] = load_frag_2x8(qrow + half * 16, qrow + half * 16 + 8);
    aq[qt][1] = load_frag_2x8(qrow + 32 + half * 16, qrow + 40 + half * 16);
  }

  v8f oacc[MT][4];
#pragma unroll
  for (int qt = 0; qt < MT; ++qt)
#pragma unroll
    for (int nt = 0; nt < 4; ++nt) oacc[qt][nt] = zero8();

  float m_r[MT], l_r[MT];               // running row stats (row = lr)
#pragma unroll
  for (int qt = 0; qt < MT; ++qt) { m_r[qt] = -3.0e38f; l_r[qt] = 0.0f; }

  for (int kb = 0; kb < SS; kb += 32) {
    // ---- prefetch next key/value tile into cache (global_prefetch_b8)
    if (kb + 32 < SS) {
      __builtin_prefetch(kbase + (size_t)(kb + 32 + lane) * DK, 0, 3);
      __builtin_prefetch(vbase + (size_t)(lane & (DV - 1)) * SS + kb + 32, 0, 3);
    }

    // ---- K fragments for keys kb..kb+31 (shared by all 4 q-tiles)
    const __bf16* kr0 = kbase + (size_t)(kb + lr) * DK + half * 16;
    const __bf16* kr1 = kbase + (size_t)(kb + 16 + lr) * DK + half * 16;
    v16bf bk00 = load_frag_16(kr0);
    v16bf bk01 = load_frag_16(kr0 + 32);
    v16bf bk10 = load_frag_16(kr1);
    v16bf bk11 = load_frag_16(kr1 + 32);

    // ---- V fragments (transposed layout -> contiguous), shared too
    const __bf16* vr = vbase + (size_t)kb + half * 16;
    v16bf bv0 = load_frag_16(vr + (size_t)(lr) * SS);
    v16bf bv1 = load_frag_16(vr + (size_t)(16 + lr) * SS);
    v16bf bv2 = load_frag_16(vr + (size_t)(32 + lr) * SS);
    v16bf bv3 = load_frag_16(vr + (size_t)(48 + lr) * SS);

#pragma unroll
    for (int qt = 0; qt < MT; ++qt) {
      // ---- scores: two 16x16 C tiles over keys kb..kb+31
      v8f s0 = zero8(), s1 = zero8();
      s0 = WMMA_BF16(aq[qt][0], bk00, s0);
      s0 = WMMA_BF16(aq[qt][1], bk01, s0);
      s1 = WMMA_BF16(aq[qt][0], bk10, s1);
      s1 = WMMA_BF16(aq[qt][1], bk11, s1);

      // ---- C-layout -> LDS (16 rows x 32 keys, fp32)
#pragma unroll
      for (int j = 0; j < 8; ++j) {
        stile[j + 8 * half][lr] = s0[j];
        stile[j + 8 * half][16 + lr] = s1[j];
      }
      __syncthreads();

      // ---- each lane picks its A-layout slice of row lr
      float sv[16];
#pragma unroll
      for (int e = 0; e < 8; ++e) {
        sv[e] = stile[lr][half * 16 + e];
        sv[8 + e] = stile[lr][half * 16 + 8 + e];
      }
      __syncthreads();

      // ---- online softmax (combine across the two 16-lane halves)
      float pmax = sv[0];
#pragma unroll
      for (int e = 1; e < 16; ++e) pmax = fmaxf(pmax, sv[e]);
      pmax = fmaxf(pmax, __shfl_xor(pmax, 16, 32));
      float mnew = fmaxf(m_r[qt], pmax);

      float psum = 0.f;
      v16bf ap;
#pragma unroll
      for (int e = 0; e < 16; ++e) {
        float pe = __expf(sv[e] - mnew);
        psum += pe;
        ap[e] = f2bf(pe);
      }
      psum += __shfl_xor(psum, 16, 32);
      float scale = __expf(m_r[qt] - mnew);
      l_r[qt] = l_r[qt] * scale + psum;
      m_r[qt] = mnew;

      // ---- rescale accumulators (C-layout rows j + 8*half)
#pragma unroll
      for (int j = 0; j < 8; ++j) {
        float sc = __shfl(scale, j + 8 * half, 32);
        oacc[qt][0][j] *= sc;
        oacc[qt][1][j] *= sc;
        oacc[qt][2][j] *= sc;
        oacc[qt][3][j] *= sc;
      }

      // ---- PV: O += P(16x32) x V(32x64)
      oacc[qt][0] = WMMA_BF16(ap, bv0, oacc[qt][0]);
      oacc[qt][1] = WMMA_BF16(ap, bv1, oacc[qt][1]);
      oacc[qt][2] = WMMA_BF16(ap, bv2, oacc[qt][2]);
      oacc[qt][3] = WMMA_BF16(ap, bv3, oacc[qt][3]);
    }
  }

  // ---- normalize and write bf16 into cat[B,S,NHEAD*DV]
#pragma unroll
  for (int qt = 0; qt < MT; ++qt) {
#pragma unroll
    for (int j = 0; j < 8; ++j) {
      float lsum = __shfl(l_r[qt], j + 8 * half, 32);
      float linv = 1.0f / lsum;
      int m = j + 8 * half;
      size_t row = (size_t)b * SS + qb * 64 + qt * 16 + m;
      __bf16* dst = cat + row * (NHEAD * DV) + h * DV + lr;
      dst[0]  = f2bf(oacc[qt][0][j] * linv);
      dst[16] = f2bf(oacc[qt][1][j] * linv);
      dst[32] = f2bf(oacc[qt][2][j] * linv);
      dst[48] = f2bf(oacc[qt][3][j] * linv);
    }
  }
}

// ---------------------------------------------------------------------------
// K3: FC projection. grid = (128 row-blocks of 64, 16 col-groups of 64).
// One wave computes a 64x64 tile. fp32 out.
// ---------------------------------------------------------------------------
__global__ __launch_bounds__(32)
void fc_kernel(const __bf16* __restrict__ cat, const __bf16* __restrict__ WfcT,
               const float* __restrict__ bfc, float* __restrict__ y) {
  const int lane = threadIdx.x;
  const int half = lane >> 4;
  const int lr = lane & 15;
  const int rb = blockIdx.x;
  const int ct = blockIdx.y;

  v8f acc[MT][4];
#pragma unroll
  for (int mt = 0; mt < MT; ++mt)
#pragma unroll
    for (int nt = 0; nt < 4; ++nt) acc[mt][nt] = zero8();

  const __bf16* arow0 = cat + (size_t)(rb * 64 + lr) * DMODEL;

  for (int k0 = 0; k0 < DMODEL; k0 += 32) {
    v16bf a[MT];
#pragma unroll
    for (int mt = 0; mt < MT; ++mt) {
      const __bf16* ar = arow0 + (size_t)mt * 16 * DMODEL + k0;
      a[mt] = load_frag_2x8(ar + half * 16, ar + half * 16 + 8);
    }
    const __bf16* wb = WfcT + (size_t)(ct * 64) * DMODEL + k0 + half * 16;
    v16bf b0 = load_frag_16(wb + (size_t)(lr) * DMODEL);
    v16bf b1 = load_frag_16(wb + (size_t)(16 + lr) * DMODEL);
    v16bf b2 = load_frag_16(wb + (size_t)(32 + lr) * DMODEL);
    v16bf b3 = load_frag_16(wb + (size_t)(48 + lr) * DMODEL);
#pragma unroll
    for (int mt = 0; mt < MT; ++mt) {
      acc[mt][0] = WMMA_BF16(a[mt], b0, acc[mt][0]);
      acc[mt][1] = WMMA_BF16(a[mt], b1, acc[mt][1]);
      acc[mt][2] = WMMA_BF16(a[mt], b2, acc[mt][2]);
      acc[mt][3] = WMMA_BF16(a[mt], b3, acc[mt][3]);
    }
  }

#pragma unroll
  for (int mt = 0; mt < MT; ++mt) {
#pragma unroll
    for (int j = 0; j < 8; ++j) {
      int m = j + 8 * half;
      size_t row = (size_t)rb * 64 + mt * 16 + m;
      float* dst = y + row * DMODEL + ct * 64 + lr;
      dst[0]  = acc[mt][0][j] + bfc[ct * 64 + lr];
      dst[16] = acc[mt][1][j] + bfc[ct * 64 + 16 + lr];
      dst[32] = acc[mt][2][j] + bfc[ct * 64 + 32 + lr];
      dst[48] = acc[mt][3][j] + bfc[ct * 64 + 48 + lr];
    }
  }
}

// ---------------------------------------------------------------------------
// K4: LayerNorm in place on y (one 256-thread block per row of 1024)
// ---------------------------------------------------------------------------
__global__ __launch_bounds__(256)
void ln_kernel(float* __restrict__ y, const float* __restrict__ gamma,
               const float* __restrict__ beta) {
  __shared__ float red[16];
  const int t = threadIdx.x;
  float* p = y + (size_t)blockIdx.x * DMODEL;

  float x0 = p[t], x1 = p[t + 256], x2 = p[t + 512], x3 = p[t + 768];
  float s = x0 + x1 + x2 + x3;
  float ss = x0 * x0 + x1 * x1 + x2 * x2 + x3 * x3;
#pragma unroll
  for (int off = 1; off < 32; off <<= 1) {
    s += __shfl_xor(s, off, 32);
    ss += __shfl_xor(ss, off, 32);
  }
  if ((t & 31) == 0) {
    red[t >> 5] = s;
    red[8 + (t >> 5)] = ss;
  }
  __syncthreads();
  float tot = 0.f, totss = 0.f;
#pragma unroll
  for (int i = 0; i < 8; ++i) { tot += red[i]; totss += red[8 + i]; }
  float mu = tot * (1.0f / DMODEL);
  float var = totss * (1.0f / DMODEL) - mu * mu;
  float inv = rsqrtf(var + 1e-5f);

  p[t]       = (x0 - mu) * inv * gamma[t]       + beta[t];
  p[t + 256] = (x1 - mu) * inv * gamma[t + 256] + beta[t + 256];
  p[t + 512] = (x2 - mu) * inv * gamma[t + 512] + beta[t + 512];
  p[t + 768] = (x3 - mu) * inv * gamma[t + 768] + beta[t + 768];
}

// ---------------------------------------------------------------------------
extern "C" void kernel_launch(void* const* d_in, const int* in_sizes, int n_in,
                              void* d_out, int out_size, void* d_ws, size_t ws_size,
                              hipStream_t stream) {
  (void)in_sizes; (void)n_in; (void)out_size; (void)ws_size;
  const float* x     = (const float*)d_in[0];
  const float* Wq    = (const float*)d_in[1];
  const float* bq    = (const float*)d_in[2];
  const float* Wk    = (const float*)d_in[3];
  const float* bk    = (const float*)d_in[4];
  const float* Wv    = (const float*)d_in[5];
  const float* bv    = (const float*)d_in[6];
  const float* Wfc   = (const float*)d_in[7];
  const float* bfc   = (const float*)d_in[8];
  const float* gamma = (const float*)d_in[9];
  const float* beta  = (const float*)d_in[10];

  // workspace layout (bytes); total ~89 MiB
  char* ws = (char*)d_ws;
  const size_t SZ_WT  = (size_t)NHEAD * DK * DMODEL * 2;   // 2 MiB each
  const size_t SZ_QKV = (size_t)BB * NHEAD * SS * DK * 2;  // 16 MiB each
  __bf16* wqT  = (__bf16*)(ws);
  __bf16* wkT  = (__bf16*)(ws + SZ_WT);
  __bf16* wvT  = (__bf16*)(ws + 2 * SZ_WT);
  __bf16* wfcT = (__bf16*)(ws + 3 * SZ_WT);
  char* base = ws + 3 * SZ_WT + (size_t)DMODEL * DMODEL * 2;
  __bf16* qbf  = (__bf16*)(base);
  __bf16* kbf  = (__bf16*)(base + SZ_QKV);
  __bf16* vTbf = (__bf16*)(base + 2 * SZ_QKV);
  __bf16* catb = (__bf16*)(base + 3 * SZ_QKV);
  __bf16* xbf  = (__bf16*)(base + 4 * SZ_QKV);

  // K0: convert x to bf16; weight transposes + bf16 convert
  {
    size_t total4 = (size_t)ROWS * DMODEL / 4;
    cvt_bf16_kernel<<<(unsigned)((total4 + 255) / 256), 256, 0, stream>>>(x, xbf, total4);
    int total = NHEAD * DMODEL * DK;
    transpose_cvt_kernel<<<(total + 255) / 256, 256, 0, stream>>>(Wq, wqT, NHEAD, DMODEL, DK);
    transpose_cvt_kernel<<<(total + 255) / 256, 256, 0, stream>>>(Wk, wkT, NHEAD, DMODEL, DK);
    transpose_cvt_kernel<<<(total + 255) / 256, 256, 0, stream>>>(Wv, wvT, NHEAD, DMODEL, DK);
    int totf = DMODEL * DMODEL;
    transpose_cvt_kernel<<<(totf + 255) / 256, 256, 0, stream>>>(Wfc, wfcT, 1, DMODEL, DMODEL);
  }

  // K1: QKV projection (WMMA, 64x64 tile per wave)
  qkv_proj_kernel<<<dim3(ROWS / 64, NHEAD, 3), 32, 0, stream>>>(
      xbf, wqT, wkT, wvT, bq, bk, bv, qbf, kbf, vTbf);

  // K2: flash attention (WMMA, 64 queries per wave)
  attn_kernel<<<dim3(SS / 64, NHEAD, BB), 32, 0, stream>>>(qbf, kbf, vTbf, catb);

  // K3: FC projection (WMMA, 64x64 tile per wave) -> fp32 y in d_out
  fc_kernel<<<dim3(ROWS / 64, DMODEL / 64), 32, 0, stream>>>(catb, wfcT, bfc, (float*)d_out);

  // K4: LayerNorm in place
  ln_kernel<<<ROWS, 256, 0, stream>>>((float*)d_out, gamma, beta);
}
